// mpnn_forward_model_21689584845244
// MI455X (gfx1250) — matrix operand order repaired and checked by reading, not compiled, not run on hardware
//
#include <hip/hip_runtime.h>
#include <stdint.h>

#define NN   20000
#define NE   640000
#define NF   16
#define EF   4
#define HIDD 128
#define LATD 128
#define OUTF 7

typedef __bf16 v16bf __attribute__((ext_vector_type(16)));
typedef float  v8f   __attribute__((ext_vector_type(8)));
typedef unsigned int v4u __attribute__((ext_vector_type(4)));

union FragU { v16bf bf; v4u u4[2]; };

__device__ __forceinline__ unsigned short f2bf(float f) {
  union { float f; unsigned int u; } c; c.f = f;
  return (unsigned short)((c.u + 0x7fffu + ((c.u >> 16) & 1u)) >> 16);
}

// A fragment (16x32 bf16, M x K) from LDS, documented gfx1250 layout:
// lanes 0-15: M=lane, elems 0..7 -> K 0..7,  elems 8..15 -> K 16..23
// lanes 16-31: M=lane-16, elems 0..7 -> K 8..15, elems 8..15 -> K 24..31
__device__ __forceinline__ v16bf load_a_frag(const unsigned short* lds, int stride,
                                             int kt, int lane) {
  int m = lane & 15;
  int koff = kt * 32 + ((lane & 16) ? 8 : 0);
  const unsigned short* p = lds + m * stride + koff;
  FragU f;
  f.u4[0] = *(const v4u*)(p);
  f.u4[1] = *(const v4u*)(p + 16);
  return f.bf;
}

// B fragment (32x16 bf16, K x N) from pre-packed weights: lane reads 32
// contiguous bytes (1KB coalesced per fragment across the wave).
__device__ __forceinline__ v16bf load_b_frag(const unsigned short* pw, int kt,
                                             int nt, int NT, int lane) {
  const v4u* p = (const v4u*)(pw + (((size_t)(kt * NT + nt) * 32 + lane) << 4));
  FragU f;
  f.u4[0] = p[0];
  f.u4[1] = p[1];
  return f.bf;
}

// Gather W fp32 columns of a segment into the wave's 16-row LDS tile as bf16.
template<int W>
__device__ __forceinline__ void stage_seg(unsigned short* A, int stride, int coff,
                                          const float* src, int srcw,
                                          const int* idx, int base, int lane) {
  constexpr int C4 = W / 4;
  for (int i = lane; i < 16 * C4; i += 32) {
    int r = i / C4;
    int c = (i % C4) * 4;
    int srow = idx ? idx[base + r] : (base + r);
    const float4 v = *(const float4*)(src + (size_t)srow * srcw + c);
    uint2 d;
    d.x = (unsigned)f2bf(v.x) | ((unsigned)f2bf(v.y) << 16);
    d.y = (unsigned)f2bf(v.z) | ((unsigned)f2bf(v.w) << 16);
    *(uint2*)(A + r * stride + coff + c) = d;
  }
}

// Full 2-layer MLP on one 16-row tile, written in the kt-outer / nt-inner
// form the backend scheduler prefers: all 8 accumulator tiles live, one A
// fragment at a time (2-deep prefetch), B fragments rotated through a 3-deep
// in-place buffer so the weight stream overlaps the matrix pipe without
// register copies. Hidden (16x128 bf16) reuses the A LDS region (DS ops from
// one wave complete in order). emit(nt, r, m, n, value).
template<int KT1, int NT2, typename F>
__device__ __forceinline__ void mlp_tile(unsigned short* lds, int strideA,
                                         const unsigned short* pw1, const float* pb1,
                                         const unsigned short* pw2, const float* pb2,
                                         int nb2, int lane, F&& emit) {
  const int nlow = lane & 15;
  const int mhi = (lane & 16) ? 8 : 0;

  // Hoist all bias loads out of the WMMA stream.
  float bias1[8];
#pragma unroll
  for (int nt = 0; nt < 8; ++nt) bias1[nt] = pb1[nt * 16 + nlow];
  float bias2[NT2];
#pragma unroll
  for (int nt = 0; nt < NT2; ++nt) {
    int n = nt * 16 + nlow;
    bias2[nt] = (n < nb2) ? pb2[n] : 0.f;
  }

  // ---- GEMM1: hidden(16x128) = relu(A x W1 + b1) ----
  constexpr int IT1 = KT1 * 8;
  v8f acc[8];
#pragma unroll
  for (int nt = 0; nt < 8; ++nt) { v8f z = {}; acc[nt] = z; }

  v16bf abuf[2];
  abuf[0] = load_a_frag(lds, strideA, 0, lane);
  v16bf bbuf[3];
#pragma unroll
  for (int i = 0; i < 3; ++i) bbuf[i] = load_b_frag(pw1, i / 8, i % 8, 8, lane);

#pragma unroll
  for (int it = 0; it < IT1; ++it) {
    const int kt = it / 8;
    const int nt = it % 8;
    if (nt == 0 && kt + 1 < KT1)
      abuf[(kt + 1) & 1] = load_a_frag(lds, strideA, kt + 1, lane);
    acc[nt] = __builtin_amdgcn_wmma_f32_16x16x32_bf16(false, abuf[kt & 1], false,
                                                      bbuf[it % 3], (short)0,
                                                      acc[nt], false, false);
    if (it + 3 < IT1)
      bbuf[it % 3] = load_b_frag(pw1, (it + 3) / 8, (it + 3) % 8, 8, lane);
  }
  asm volatile("s_wait_dscnt 0" ::: "memory");  // A reads done before overwrite

#pragma unroll
  for (int nt = 0; nt < 8; ++nt) {
#pragma unroll
    for (int r = 0; r < 8; ++r) {  // C layout: reg r -> M=r+mhi, N=nt*16+nlow
      float h = fmaxf(acc[nt][r] + bias1[nt], 0.f);
      lds[(r + mhi) * HIDD + nt * 16 + nlow] = f2bf(h);
    }
  }
  asm volatile("s_wait_dscnt 0" ::: "memory");  // hidden visible wave-wide

  // ---- GEMM2: out(16 x NT2*16) = hidden x W2 + b2 ----
  constexpr int IT2 = 4 * NT2;
  v8f acc2[NT2];
#pragma unroll
  for (int nt = 0; nt < NT2; ++nt) { v8f z = {}; acc2[nt] = z; }

  v16bf hbuf[2];
  hbuf[0] = load_a_frag(lds, HIDD, 0, lane);
  v16bf cbuf[3];
#pragma unroll
  for (int i = 0; i < 3; ++i) cbuf[i] = load_b_frag(pw2, i / NT2, i % NT2, NT2, lane);

#pragma unroll
  for (int it = 0; it < IT2; ++it) {
    const int kt = it / NT2;
    const int nt = it % NT2;
    if (nt == 0 && kt + 1 < 4)
      hbuf[(kt + 1) & 1] = load_a_frag(lds, HIDD, kt + 1, lane);
    acc2[nt] = __builtin_amdgcn_wmma_f32_16x16x32_bf16(false, hbuf[kt & 1], false,
                                                       cbuf[it % 3], (short)0,
                                                       acc2[nt], false, false);
    if (it + 3 < IT2)
      cbuf[it % 3] = load_b_frag(pw2, (it + 3) / NT2, (it + 3) % NT2, NT2, lane);
  }

#pragma unroll
  for (int nt = 0; nt < NT2; ++nt) {
    int n = nt * 16 + nlow;
#pragma unroll
    for (int r = 0; r < 8; ++r) emit(nt, r, r + mhi, n, acc2[nt][r] + bias2[nt]);
  }
}

// MODE: 0=encoder edge (K=36->64), 1=processor edge (K=384, residual),
//       2=decoder edge (K=384, scatter only)
template<int MODE>
__global__ __launch_bounds__(128) void edge_kernel(
    const float* __restrict__ nodef, const float* efeat,
    const int* __restrict__ erow, const int* __restrict__ ecol,
    const unsigned short* __restrict__ pw1, const float* __restrict__ b1,
    const unsigned short* __restrict__ pw2, const float* __restrict__ b2,
    float* elbuf, float* agg) {
  constexpr int KP = (MODE == 0) ? 64 : 384;
  constexpr int KT = KP / 32;
  constexpr int LW = (KP > 128) ? KP : 128;
  __shared__ __align__(16) unsigned short lds[4][16][LW];
  const int wave = threadIdx.x >> 5;
  const int lane = threadIdx.x & 31;
  unsigned short* A = &lds[wave][0][0];
  const int base = (blockIdx.x * 4 + wave) * 16;  // NE = 10000*64 exactly

  if constexpr (MODE == 0) {
    for (int i = lane; i < 16 * 28; i += 32) {  // zero pad cols [36,64)
      int r = i / 28, c = 36 + i % 28;
      A[r * LW + c] = 0;
    }
    stage_seg<16>(A, LW, 0,  nodef, NF, erow, base, lane);
    stage_seg<16>(A, LW, 16, nodef, NF, ecol, base, lane);
    stage_seg<4>(A, LW, 32, efeat, EF, nullptr, base, lane);
  } else {
    stage_seg<128>(A, LW, 0,   nodef, HIDD, erow, base, lane);
    stage_seg<128>(A, LW, 128, nodef, HIDD, ecol, base, lane);
    stage_seg<128>(A, LW, 256, efeat, LATD, nullptr, base, lane);
  }
  asm volatile("s_wait_dscnt 0" ::: "memory");

  const int mhi = (lane & 16) ? 8 : 0;
  int cols[8];
#pragma unroll
  for (int r = 0; r < 8; ++r) cols[r] = ecol[base + r + mhi];

  mlp_tile<KT, 8>(A, LW, pw1, b1, pw2, b2, 128, lane,
    [&](int nt, int r, int m, int n, float v) {
      (void)nt;
      atomicAdd(agg + (size_t)cols[r] * LATD + n, v);  // segment_sum(e_new)
      if constexpr (MODE == 0) {
        elbuf[(size_t)(base + m) * LATD + n] = v;
      } else if constexpr (MODE == 1) {
        elbuf[(size_t)(base + m) * LATD + n] += v;     // el += e_new
      }
    });
}

// MODE: 0=encoder node (K=144->160), 1=processor node (K=256, residual),
//       2=decoder node (K=256, out 7)
template<int MODE>
__global__ __launch_bounds__(128) void node_kernel(
    const float* feat, const float* __restrict__ agg,
    const unsigned short* __restrict__ pw1, const float* __restrict__ b1,
    const unsigned short* __restrict__ pw2, const float* __restrict__ b2,
    float* outp) {
  constexpr int KP = (MODE == 0) ? 160 : 256;
  constexpr int KT = KP / 32;
  constexpr int NT2 = (MODE == 2) ? 1 : 8;
  __shared__ __align__(16) unsigned short lds[4][16][KP];
  const int wave = threadIdx.x >> 5;
  const int lane = threadIdx.x & 31;
  const int tile = blockIdx.x * 4 + wave;
  if (tile >= NN / 16) return;  // wave-uniform; no barriers used below
  unsigned short* A = &lds[wave][0][0];
  const int base = tile * 16;

  if constexpr (MODE == 0) {
    for (int i = lane; i < 16 * 16; i += 32) {  // zero pad cols [144,160)
      int r = i >> 4, c = 144 + (i & 15);
      A[r * KP + c] = 0;
    }
    stage_seg<16>(A, KP, 0, feat, NF, nullptr, base, lane);
    stage_seg<128>(A, KP, 16, agg, LATD, nullptr, base, lane);
  } else {
    stage_seg<128>(A, KP, 0, feat, HIDD, nullptr, base, lane);
    stage_seg<128>(A, KP, 128, agg, LATD, nullptr, base, lane);
  }
  asm volatile("s_wait_dscnt 0" ::: "memory");

  mlp_tile<KT, NT2>(A, KP, pw1, b1, pw2, b2, (MODE == 2) ? OUTF : 128, lane,
    [&](int nt, int r, int m, int n, float v) {
      (void)nt; (void)r;
      if constexpr (MODE == 0) {
        outp[(size_t)(base + m) * HIDD + n] = v;
      } else if constexpr (MODE == 1) {
        outp[(size_t)(base + m) * HIDD + n] += v;      // xl += xn
      } else {
        if (n < OUTF) outp[(size_t)(base + m) * OUTF + n] = v;
      }
    });
}

// Pack fp32 weights (Ksrc x Nsrc) into per-lane bf16 B-fragment layout.
__global__ void pack_w(const float* __restrict__ W, unsigned short* __restrict__ dst,
                       int Ksrc, int Nsrc, int KT, int NT) {
  int idx = blockIdx.x * 256 + threadIdx.x;
  int total = KT * NT * 512;
  if (idx >= total) return;
  int e = idx & 15;
  int lane = (idx >> 4) & 31;
  int rem = idx >> 9;
  int nt = rem % NT;
  int kt = rem / NT;
  int n = nt * 16 + (lane & 15);
  int k = kt * 32 + ((lane & 16) ? 8 : 0) + ((e & 8) ? 16 : 0) + (e & 7);
  float v = (k < Ksrc && n < Nsrc) ? W[(size_t)k * Nsrc + n] : 0.f;
  dst[idx] = f2bf(v);
}

__global__ void zero_f32(float* __restrict__ p, int n) {
  int i = blockIdx.x * 256 + threadIdx.x;
  if (i < n) p[i] = 0.f;
}

extern "C" void kernel_launch(void* const* d_in, const int* in_sizes, int n_in,
                              void* d_out, int out_size, void* d_ws, size_t ws_size,
                              hipStream_t stream) {
  (void)in_sizes; (void)n_in; (void)out_size; (void)ws_size;
  const float* x     = (const float*)d_in[0];
  const int*   ei    = (const int*)d_in[1];
  const float* eattr = (const float*)d_in[2];
  const float* W[6][4];  // [enc_e, enc_n, proc_e, proc_n, dec_e, dec_n][w1,b1,w2,b2]
  for (int i = 0; i < 6; ++i)
    for (int j = 0; j < 4; ++j) W[i][j] = (const float*)d_in[3 + i * 4 + j];

  const int* erow = ei;
  const int* ecol = ei + NE;

  // Workspace: el (328MB) | agg (10MB) | xl (10MB) | packed bf16 weights (~0.55MB)
  float* el  = (float*)d_ws;
  float* agg = el + (size_t)NE * LATD;
  float* xl  = agg + (size_t)NN * LATD;
  unsigned short* pwbase = (unsigned short*)(xl + (size_t)NN * HIDD);

  const int din[6]  = {2 * NF + EF, NF + LATD, 2 * HIDD + LATD, HIDD + LATD,
                       2 * HIDD + LATD, HIDD + LATD};         // 36,144,384,256,384,256
  const int kt1[6]  = {2, 5, 12, 8, 12, 8};                   // ceil(din/32)
  const int dout[6] = {LATD, HIDD, LATD, HIDD, LATD, OUTF};
  const int nt2[6]  = {8, 8, 8, 8, 8, 1};
  unsigned short* pw1[6];
  unsigned short* pw2[6];
  size_t off = 0;
  for (int i = 0; i < 6; ++i) {
    pw1[i] = pwbase + off; off += (size_t)kt1[i] * 8 * 512;
    pw2[i] = pwbase + off; off += (size_t)4 * nt2[i] * 512;
  }
  for (int i = 0; i < 6; ++i) {
    int t1 = kt1[i] * 8 * 512;
    pack_w<<<dim3((t1 + 255) / 256), dim3(256), 0, stream>>>(
        W[i][0], pw1[i], din[i], HIDD, kt1[i], 8);
    int t2 = 4 * nt2[i] * 512;
    pack_w<<<dim3((t2 + 255) / 256), dim3(256), 0, stream>>>(
        W[i][2], pw2[i], HIDD, dout[i], 4, nt2[i]);
  }

  const int aggN = NN * LATD;
  dim3 zgrid((aggN + 255) / 256);
  dim3 egrid(NE / 64);                 // 10000 blocks, 4 waves x 16 edges
  dim3 ngrid((NN / 16 + 3) / 4);       // 313 blocks (1250 wave-tiles)
  dim3 blk(128);

  // encoder
  zero_f32<<<zgrid, 256, 0, stream>>>(agg, aggN);
  edge_kernel<0><<<egrid, blk, 0, stream>>>(x, eattr, erow, ecol,
                                            pw1[0], W[0][1], pw2[0], W[0][3], el, agg);
  node_kernel<0><<<ngrid, blk, 0, stream>>>(x, agg, pw1[1], W[1][1], pw2[1], W[1][3], xl);
  // processor rounds (shared params, residual)
  for (int rr = 0; rr < 4; ++rr) {
    zero_f32<<<zgrid, 256, 0, stream>>>(agg, aggN);
    edge_kernel<1><<<egrid, blk, 0, stream>>>(xl, el, erow, ecol,
                                              pw1[2], W[2][1], pw2[2], W[2][3], el, agg);
    node_kernel<1><<<ngrid, blk, 0, stream>>>(xl, agg, pw1[3], W[3][1], pw2[3], W[3][3], xl);
  }
  // decoder
  zero_f32<<<zgrid, 256, 0, stream>>>(agg, aggN);
  edge_kernel<2><<<egrid, blk, 0, stream>>>(xl, el, erow, ecol,
                                            pw1[4], W[4][1], pw2[4], W[4][3], nullptr, agg);
  node_kernel<2><<<ngrid, blk, 0, stream>>>(xl, agg, pw1[5], W[5][1], pw2[5], W[5][3],
                                            (float*)d_out);
}
